// BiAttentionLayer_23751169147270
// MI455X (gfx1250) — matrix operand order repaired and checked by reading, not compiled, not run on hardware
//
#include <hip/hip_runtime.h>

// ---------------------------------------------------------------------------
// BiAttention (BiDAF) fused flash kernel for MI455X / gfx1250 (wave32, WMMA)
//   S[t,j] = qw[j] + (c[t]*wm) . q[j]          (wc term is softmax-invariant)
//   A = softmax_j(S);  U = A @ q;  b_t = 1/l_t; h = sum_t b_t*c[t]
//   G = [c, U, c*U, c*h]   (8192 x 400 f32)
// Tricks: qw folded into the S-GEMM via a 1.0 column at k=100;
//         softmax denominator computed by the U-GEMM via a ones N-tile.
// ---------------------------------------------------------------------------

typedef __attribute__((ext_vector_type(16))) __bf16 v16bf;
typedef __attribute__((ext_vector_type(8)))  __bf16 v8bf;
typedef __attribute__((ext_vector_type(8)))  float  v8f;
typedef __attribute__((ext_vector_type(4)))  float  v4f;

#define T_N   8192
#define J_N   8192
#define D_N   100
#define JT    64      // J tile: 4 S-subtiles, U-GEMM K = 64 (2 wmma K-steps)
#define KP    128     // D padded: k=100 carries the qw ones-column
#define NP2   128     // U-GEMM N: 0..99 data, 100..111 zero, 112..127 ones (=rowsum)
#define NTL2  8
#define WAVES 8
#define BLK_T 128     // 8 waves x 16 rows
#define PSTR  72      // P staging stride (bf16): 36-dword rows, conflict-free
#define OUTW  400

__device__ __forceinline__ unsigned pack2bf(float lo, float hi) {
  unsigned short a = __builtin_bit_cast(unsigned short, (__bf16)lo);
  unsigned short b = __builtin_bit_cast(unsigned short, (__bf16)hi);
  return (unsigned)a | ((unsigned)b << 16);
}

// qw[j] = q[j,:] . kernel[D:2D]
__global__ void qw_kernel(const float* __restrict__ q,
                          const float* __restrict__ kern,
                          float* __restrict__ qw) {
  int j = blockIdx.x * blockDim.x + threadIdx.x;
  if (j >= J_N) return;
  const float* wq = kern + D_N;
  float s = 0.f;
  for (int d = 0; d < D_N; ++d) s += q[j * D_N + d] * wq[d];
  qw[j] = s;
}

__global__ __launch_bounds__(256, 1)
void flash_kernel(const float* __restrict__ c, const float* __restrict__ q,
                  const float* __restrict__ kern, const float* __restrict__ qw,
                  float* __restrict__ out, float* __restrict__ binv) {
  __shared__ __attribute__((aligned(32))) __bf16 ldsq[JT * KP];    // q rows + qw col
  __shared__ __attribute__((aligned(32))) __bf16 ldsqt[NP2 * JT];  // q^T + ones cols
  __shared__ __attribute__((aligned(16))) __bf16 ldsP[WAVES][16 * PSTR];

  const int tid  = threadIdx.x;
  const int wave = tid >> 5;
  const int lane = tid & 31;
  const int half = lane >> 4;     // C-layout half (rows +8)
  const int l16  = lane & 15;
  const int t0   = blockIdx.x * BLK_T + wave * 16;
  int xa[4];                      // hoisted bpermute byte addresses
#pragma unroll
  for (int m = 0; m < 4; ++m) xa[m] = ((lane ^ (1 << m)) << 2);

  // ---- one-time static LDS padding: ldsq cols 100..127 zero (100 rewritten
  //      per tile with qw); ldsqt cols 100..111 zero, 112..127 one (rowsum B)
  for (int i = tid; i < JT * 28; i += 256) {
    int row = i / 28, col = 100 + i % 28;
    ldsq[row * KP + col]  = (__bf16)0.f;
    ldsqt[col * JT + row] = (col < 112) ? (__bf16)0.f : (__bf16)1.f;
  }

  // ---- A fragments: [c*wm | 1 | 0pad] 16 x 128(bf16), reused for all tiles
  const float* crow = c + (t0 + l16) * D_N;
  const float* wm   = kern + 2 * D_N;
  v16bf aC[4];
#pragma unroll
  for (int s = 0; s < 4; ++s) {
    int kb = s * 32 + half * 16;  // contiguous A-layout: half selects K {0..15}/{16..31}
#pragma unroll
    for (int i = 0; i < 8; ++i) {
      int k0 = kb + i, k1 = kb + 8 + i;
      float v0 = k0 < D_N ? crow[k0] * wm[k0] : (k0 == D_N ? 1.f : 0.f);
      float v1 = k1 < D_N ? crow[k1] * wm[k1] : (k1 == D_N ? 1.f : 0.f);
      aC[s][i]     = (__bf16)v0;
      aC[s][8 + i] = (__bf16)v1;
    }
  }

  v8f U[NTL2];                    // U[7] accumulates the softmax denominator
#pragma unroll
  for (int n = 0; n < NTL2; ++n)
#pragma unroll
    for (int r = 0; r < 8; ++r) U[n][r] = 0.f;

  float m_[8];
#pragma unroll
  for (int r = 0; r < 8; ++r) m_[r] = -1.0e30f;

  for (int j0 = 0; j0 < J_N; j0 += JT) {
    __syncthreads();              // previous tile fully consumed
    {                             // cooperative q-tile load (both LDS layouts)
      int row = tid >> 2;         // 0..63
      int cbi = tid & 3;
      int cb  = cbi * 32;         // 0,32,64,96
      const float* qrow = q + (size_t)(j0 + row) * D_N;
      if (j0 + JT < J_N)          // warm L2 for next tile
        __builtin_prefetch(q + (size_t)(j0 + JT + row) * D_N + cb, 0, 1);
      int nv = (cbi < 3) ? 8 : 1; // #full float4s below col 100
#pragma unroll
      for (int w = 0; w < 8; ++w) {
        if (w < nv) {
          v4f v = *(const v4f*)(qrow + cb + w * 4);
          int cc = cb + w * 4;
          *(unsigned*)&ldsq[row * KP + cc]     = pack2bf(v.x, v.y);
          *(unsigned*)&ldsq[row * KP + cc + 2] = pack2bf(v.z, v.w);
          ldsqt[(cc + 0) * JT + row] = (__bf16)v.x;
          ldsqt[(cc + 1) * JT + row] = (__bf16)v.y;
          ldsqt[(cc + 2) * JT + row] = (__bf16)v.z;
          ldsqt[(cc + 3) * JT + row] = (__bf16)v.w;
        }
      }
      if (cbi == 3) ldsq[row * KP + 100] = (__bf16)qw[j0 + row];
    }
    __syncthreads();

    // ---- S tile (16x64) = [cm|1] @ [q|qw]^T : 4 subtiles x 4 K-steps
    v8f S[4];
#pragma unroll
    for (int g = 0; g < 4; ++g)
#pragma unroll
      for (int r = 0; r < 8; ++r) S[g][r] = 0.f;
#pragma unroll
    for (int s = 0; s < 4; ++s) {
      v16bf b0 = *(const v16bf*)&ldsq[(0 * 16 + l16) * KP + s * 32 + half * 16];
      v16bf b1 = *(const v16bf*)&ldsq[(1 * 16 + l16) * KP + s * 32 + half * 16];
      v16bf b2 = *(const v16bf*)&ldsq[(2 * 16 + l16) * KP + s * 32 + half * 16];
      v16bf b3 = *(const v16bf*)&ldsq[(3 * 16 + l16) * KP + s * 32 + half * 16];
      S[0] = __builtin_amdgcn_wmma_f32_16x16x32_bf16(false, aC[s], false, b0,
                                                     (short)0, S[0], false, false);
      S[1] = __builtin_amdgcn_wmma_f32_16x16x32_bf16(false, aC[s], false, b1,
                                                     (short)0, S[1], false, false);
      S[2] = __builtin_amdgcn_wmma_f32_16x16x32_bf16(false, aC[s], false, b2,
                                                     (short)0, S[2], false, false);
      S[3] = __builtin_amdgcn_wmma_f32_16x16x32_bf16(false, aC[s], false, b3,
                                                     (short)0, S[3], false, false);
    }

    // ---- online softmax: batched 16-lane max reduction (8 rows per step)
    float red[8], tmp[8];
#pragma unroll
    for (int r = 0; r < 8; ++r)
      red[r] = fmaxf(fmaxf(S[0][r], S[1][r]), fmaxf(S[2][r], S[3][r]));
#pragma unroll
    for (int m = 0; m < 4; ++m) {
#pragma unroll
      for (int r = 0; r < 8; ++r)
        tmp[r] = __builtin_bit_cast(float,
                   __builtin_amdgcn_ds_bpermute(xa[m],
                     __builtin_bit_cast(int, red[r])));
#pragma unroll
      for (int r = 0; r < 8; ++r) red[r] = fmaxf(red[r], tmp[r]);
    }
    float scale[8];
#pragma unroll
    for (int r = 0; r < 8; ++r) {
      float mn = fmaxf(m_[r], red[r]);
      scale[r] = __expf(m_[r] - mn);
      m_[r] = mn;
#pragma unroll
      for (int g = 0; g < 4; ++g) S[g][r] = __expf(S[g][r] - mn);
    }
#pragma unroll
    for (int n = 0; n < NTL2; ++n)      // rescale U and the denominator tile
#pragma unroll
      for (int r = 0; r < 8; ++r) U[n][r] *= scale[r];

    // ---- C-layout -> A-layout transpose of P through per-wave LDS scratch
    __bf16* Pw = ldsP[wave];
#pragma unroll
    for (int r = 0; r < 8; ++r) {
      int row = r + half * 8;
#pragma unroll
      for (int g = 0; g < 4; ++g)
        Pw[row * PSTR + g * 16 + l16] = (__bf16)S[g][r];
    }
    int pkb = half * 16;
    v16bf pA[2];
#pragma unroll
    for (int u = 0; u < 2; ++u) {
      v8bf plo = *(const v8bf*)&Pw[l16 * PSTR + u * 32 + pkb];
      v8bf phi = *(const v8bf*)&Pw[l16 * PSTR + u * 32 + pkb + 8];
      pA[u] = __builtin_shufflevector(plo, phi, 0, 1, 2, 3, 4, 5, 6, 7,
                                      8, 9, 10, 11, 12, 13, 14, 15);
    }

    // ---- [U | l] (16x128) += P(16x64) @ [q_tile | 0 | 1] : 8 N x 2 K wmmas
#pragma unroll
    for (int n = 0; n < NTL2; ++n) {
      int dcol = n * 16 + l16;
      v16bf qb0 = *(const v16bf*)&ldsqt[dcol * JT + 0 * 32 + half * 16];
      v16bf qb1 = *(const v16bf*)&ldsqt[dcol * JT + 1 * 32 + half * 16];
      U[n] = __builtin_amdgcn_wmma_f32_16x16x32_bf16(false, pA[0], false, qb0,
                                                     (short)0, U[n], false, false);
      U[n] = __builtin_amdgcn_wmma_f32_16x16x32_bf16(false, pA[1], false, qb1,
                                                     (short)0, U[n], false, false);
    }
  }

  // ---- epilogue: U/l into out[:,100:200], b = 1/l into workspace
#pragma unroll
  for (int r = 0; r < 8; ++r) {
    float inv = 1.0f / U[7][r];   // ones-tile column = softmax denominator
    int t = t0 + r + half * 8;
    if (l16 == 0) binv[t] = inv;
#pragma unroll
    for (int n = 0; n < 7; ++n) {
      int d = n * 16 + l16;
      if (d < D_N) out[(size_t)t * OUTW + 100 + d] = U[n][r] * inv;
    }
  }
}

// h[d] = sum_t b_t * c[t,d] -- single block, fixed order => deterministic
__global__ void hred_kernel(const float* __restrict__ c,
                            const float* __restrict__ binv,
                            float* __restrict__ h) {
  int d = threadIdx.x;
  if (d >= D_N) return;
  float acc = 0.f;
  for (int t = 0; t < T_N; ++t) acc += binv[t] * c[(size_t)t * D_N + d];
  h[d] = acc;
}

__global__ void fin_kernel(const float* __restrict__ c,
                           const float* __restrict__ h,
                           float* __restrict__ out) {
  int idx = blockIdx.x * blockDim.x + threadIdx.x;
  if (idx >= T_N * D_N) return;
  int t = idx / D_N, d = idx - t * D_N;
  float cv = c[idx];
  float uv = out[(size_t)t * OUTW + 100 + d];
  out[(size_t)t * OUTW + d]       = cv;
  out[(size_t)t * OUTW + 200 + d] = cv * uv;
  out[(size_t)t * OUTW + 300 + d] = cv * h[d];
}

extern "C" void kernel_launch(void* const* d_in, const int* in_sizes, int n_in,
                              void* d_out, int out_size, void* d_ws, size_t ws_size,
                              hipStream_t stream) {
  (void)in_sizes; (void)n_in; (void)out_size; (void)ws_size;
  const float* c    = (const float*)d_in[0];   // context  (1,8192,100)
  const float* q    = (const float*)d_in[1];   // question (1,8192,100)
  const float* kern = (const float*)d_in[2];   // (300,)
  float* out = (float*)d_out;                  // (8192,400)

  float* w    = (float*)d_ws;
  float* qw   = w;               // 8192 f32
  float* binv = w + J_N;         // 8192 f32
  float* h    = w + 2 * J_N;     // 128  f32

  qw_kernel<<<J_N / 256, 256, 0, stream>>>(q, kern, qw);
  flash_kernel<<<T_N / BLK_T, 256, 0, stream>>>(c, q, kern, qw, out, binv);
  hred_kernel<<<1, 128, 0, stream>>>(c, binv, h);
  fin_kernel<<<(T_N * D_N) / 256, 256, 0, stream>>>(c, h, out);
}
